// ScaffoldGenerator_7146825581067
// MI455X (gfx1250) — matrix operand order repaired and checked by reading, not compiled, not run on hardware
//
#include <hip/hip_runtime.h>

// ---------------------------------------------------------------------------
// CDNA5 (gfx1250) EGNN scaffold-generator forward pass.
// B=2, N=256, H=256, D=64, L=6.  All heavy GEMMs use v_wmma_f32_16x16x32_bf16
// (wave32, fp32 accumulate).  Pairwise edge MLP is fused per 16x16 pair tile.
// ---------------------------------------------------------------------------

typedef __bf16 bf16x16 __attribute__((ext_vector_type(16)));
typedef __bf16 bf16x8  __attribute__((ext_vector_type(8)));
typedef float  f32x8   __attribute__((ext_vector_type(8)));

#define H_DIM 256
#define NN    256
#define BB    2
#define ROWS  (BB * NN)        // 512 node rows
#define LAYERS 6

__device__ __forceinline__ unsigned short f2bf(float f) {
  unsigned int u = __float_as_uint(f);
  unsigned int r = (u + 0x7FFFu + ((u >> 16) & 1u)) >> 16;   // round-nearest-even
  return (unsigned short)r;
}
__device__ __forceinline__ float bf2f(unsigned short h) {
  return __uint_as_float(((unsigned int)h) << 16);
}
__device__ __forceinline__ float silu_f(float x) { return x / (1.f + __expf(-x)); }

__device__ __forceinline__ f32x8 wmma_bf16(bf16x16 a, bf16x16 b, f32x8 c) {
  // args: (neg_a, A, neg_b, B, c_mod, C, reuse_a, reuse_b)
  return __builtin_amdgcn_wmma_f32_16x16x32_bf16(false, a, false, b, (short)0, c,
                                                 false, false);
}

// ---------------------------------------------------------------------------
// Weight prep: fp32 [K=256][N=256] -> bf16 transposed [N][K] so every WMMA
// B-fragment (16 contiguous K values for one output column) is a single
// 32-byte global load.
// ---------------------------------------------------------------------------
__global__ void ct_kernel(const float* __restrict__ src, unsigned short* __restrict__ dst) {
  int n = blockIdx.x, k = threadIdx.x;
  dst[n * 256 + k] = f2bf(src[k * 256 + n]);
}

// ---------------------------------------------------------------------------
// Pocket encoder + time embedding.  One block per node row; tiny FLOPs.
// Also initializes x = pocket_coords.
// ---------------------------------------------------------------------------
__global__ __launch_bounds__(256) void enc_kernel(
    const float* __restrict__ coords, const float* __restrict__ feats,
    const float* __restrict__ t,
    const float* __restrict__ w1, const float* __restrict__ b1,
    const float* __restrict__ w2, const float* __restrict__ b2,
    const float* __restrict__ w3, const float* __restrict__ b3,
    const float* __restrict__ tw1, const float* __restrict__ tb1,
    const float* __restrict__ tw2, const float* __restrict__ tb2,
    float* __restrict__ h, float* __restrict__ x) {
  int row = blockIdx.x, tid = threadIdx.x;
  int b = row >> 8;
  __shared__ float pin[68];
  __shared__ float bufA[256];
  __shared__ float bufB[256];
  __shared__ float te1[256];
  if (tid < 3) { pin[tid] = coords[row * 3 + tid]; x[row * 3 + tid] = coords[row * 3 + tid]; }
  if (tid < 64) pin[3 + tid] = feats[row * 64 + tid];
  te1[tid] = fmaxf(t[b] * tw1[tid] + tb1[tid], 0.f);
  __syncthreads();
  float acc = b1[tid];
  for (int i = 0; i < 67; ++i) acc += pin[i] * w1[i * 256 + tid];
  bufA[tid] = fmaxf(acc, 0.f);
  __syncthreads();
  acc = b2[tid];
  for (int i = 0; i < 256; ++i) acc += bufA[i] * w2[i * 256 + tid];
  bufB[tid] = fmaxf(acc, 0.f);
  __syncthreads();
  acc = b3[tid];
  for (int i = 0; i < 256; ++i) acc += bufB[i] * w3[i * 256 + tid];
  float h3 = fmaxf(acc, 0.f);
  acc = tb2[tid];
  for (int i = 0; i < 256; ++i) acc += te1[i] * tw2[i * 256 + tid];
  h[row * 256 + tid] = h3 + acc;
}

// ---------------------------------------------------------------------------
// LayerNorm (per row over H=256): hn = (h-mu)/sqrt(var+1e-5)*g + b
// ---------------------------------------------------------------------------
__global__ __launch_bounds__(256) void ln_kernel(const float* __restrict__ h,
                                                 const float* __restrict__ g,
                                                 const float* __restrict__ beta,
                                                 float* __restrict__ hn) {
  int row = blockIdx.x, tid = threadIdx.x;
  __shared__ float red[256];
  float v = h[row * 256 + tid];
  red[tid] = v;
  __syncthreads();
  for (int s = 128; s > 0; s >>= 1) { if (tid < s) red[tid] += red[tid + s]; __syncthreads(); }
  float mu = red[0] * (1.f / 256.f);
  __syncthreads();
  float c = v - mu;
  red[tid] = c * c;
  __syncthreads();
  for (int s = 128; s > 0; s >>= 1) { if (tid < s) red[tid] += red[tid + s]; __syncthreads(); }
  float inv = rsqrtf(red[0] * (1.f / 256.f) + 1e-5f);
  hn[row * 256 + tid] = c * inv * g[tid] + beta[tid];
}

// ---------------------------------------------------------------------------
// Generic node GEMM: out[512x256] = post( A[512x256] @ W + bias + addIn ) + addPost
// A is fp32 (converted to bf16 in-register), W is bf16-transposed [N][K].
// ACT: 0=none 1=relu 2=silu; HB/HAI/HAP are compile-time post-op flags so the
// epilogue is straight-line (no per-element SCC branches / hazard stalls).
// One block per 16-row M-tile; 8 waves cover 16 N-tiles; K = 8 x (16x16x32).
// All 8 B-fragments of an N-tile are preloaded (one clause of b128 loads) so
// the 8-deep WMMA chain overlaps the load latency.
// ---------------------------------------------------------------------------
template <int ACT, bool HB, bool HAI, bool HAP>
__global__ __launch_bounds__(256) void gemm_kernel(
    const float* __restrict__ A, const unsigned short* __restrict__ Wt,
    const float* __restrict__ bias, const float* __restrict__ addIn,
    const float* __restrict__ addPost, float* __restrict__ out) {
  int mtile = blockIdx.x, tid = threadIdx.x;
  int wave = tid >> 5, lane = tid & 31, half = lane >> 4, l16 = lane & 15;
  int mrow = mtile * 16 + l16;
  const float* arow = A + mrow * 256;

  // A fragments: lane half h holds contiguous K = 16h..16h+15 (matches B)
  bf16x16 afr[8];
#pragma unroll
  for (int kk = 0; kk < 8; ++kk) {
    f32x8 lo = *(const f32x8*)(arow + kk * 32 + half * 16);
    f32x8 hi = *(const f32x8*)(arow + kk * 32 + half * 16 + 8);
    bf16x16 a;
#pragma unroll
    for (int e = 0; e < 8; ++e) {
      a[e]     = __builtin_bit_cast(__bf16, f2bf(lo[e]));
      a[e + 8] = __builtin_bit_cast(__bf16, f2bf(hi[e]));
    }
    afr[kk] = a;
  }

  for (int nt = wave; nt < 16; nt += 8) {
    int ncol = nt * 16 + l16;
    const unsigned short* wrow = Wt + ncol * 256;
    if (nt + 8 < 16)  // warm L2/WGP$ for this wave's next N-tile
      __builtin_prefetch(Wt + (ncol + 128) * 256, 0, 1);
    bf16x16 bfr[8];
#pragma unroll
    for (int kk = 0; kk < 8; ++kk)
      bfr[kk] = *(const bf16x16*)(wrow + kk * 32 + half * 16);
    f32x8 acc = {};
#pragma unroll
    for (int kk = 0; kk < 8; ++kk) acc = wmma_bf16(afr[kk], bfr[kk], acc);
#pragma unroll
    for (int r = 0; r < 8; ++r) {
      int rg = mtile * 16 + r + 8 * half;      // C/D layout: VGPR r -> M = r + 8*half
      float v = acc[r];
      if (HB)  v += bias[ncol];
      if (HAI) v += addIn[rg * 256 + ncol];
      if (ACT == 1)      v = fmaxf(v, 0.f);
      else if (ACT == 2) v = silu_f(v);
      if (HAP) v += addPost[rg * 256 + ncol];
      out[rg * 256 + ncol] = v;
    }
  }
}

// ---------------------------------------------------------------------------
// Fused all-pairs edge kernel.  Block = (i-tile, j-tile, batch): 256 pair rows.
//   m1 = silu(A_i + Bm_j + d2*w1c)           (elementwise, -> LDS bf16)
//   m  = silu(m1 @ ew2 + eb2)                (WMMA, overwrites m1 in place)
//   c  = sum_n silu((m @ cw1)_n + cb1_n)*cw2_n + cb2   (WMMA + folded reduce)
//   magg_i += sum_j m ; xacc_i += sum_j (x_i-x_j)*c    (global fp32 atomics)
// ---------------------------------------------------------------------------
#define ESTR 272  // ushort row stride (544 B, 16B aligned, padded vs 256)

__global__ __launch_bounds__(256) void edge_kernel(
    const float* __restrict__ Aacts, const float* __restrict__ Bacts,
    const float* __restrict__ x, const float* __restrict__ w1c,
    const unsigned short* __restrict__ ew2T, const float* __restrict__ eb2,
    const unsigned short* __restrict__ cw1T, const float* __restrict__ cb1,
    const float* __restrict__ cw2, const float* __restrict__ cb2p,
    float* __restrict__ magg, float* __restrict__ xacc) {
  extern __shared__ char smem[];
  unsigned short* mbuf = (unsigned short*)smem;                 // 256 x ESTR bf16
  float* cacc = (float*)(smem + 256 * ESTR * 2);                // 256 f32
  float* d2s  = cacc + 256;                                     // 256 f32
  float* xi   = d2s + 256;                                      // 16x3
  float* xj   = xi + 48;                                        // 16x3

  int tid = threadIdx.x;
  int it = blockIdx.x, jt = blockIdx.y, b = blockIdx.z;
  int rowI = b * 256 + it * 16;
  int rowJ = b * 256 + jt * 16;

  if (tid < 48) xi[tid] = x[(rowI + tid / 3) * 3 + (tid % 3)];
  else if (tid < 96) { int u = tid - 48; xj[u] = x[(rowJ + u / 3) * 3 + (u % 3)]; }
  cacc[tid] = 0.f;
  __syncthreads();
  {
    int i = tid >> 4, j = tid & 15;
    float dx = xi[i * 3] - xj[j * 3];
    float dy = xi[i * 3 + 1] - xj[j * 3 + 1];
    float dz = xi[i * 3 + 2] - xj[j * 3 + 2];
    d2s[tid] = dx * dx + dy * dy + dz * dz;
  }
  __syncthreads();

  // ---- m1 elementwise (col = tid, loop over 256 pair rows)
  float w1cc = w1c[tid];
  for (int p = 0; p < 256; ++p) {
    int i = p >> 4, j = p & 15;
    float v = Aacts[(rowI + i) * 256 + tid] + Bacts[(rowJ + j) * 256 + tid] + d2s[p] * w1cc;
    mbuf[p * ESTR + tid] = f2bf(silu_f(v));
  }
  __syncthreads();

  // ---- two fused 256x256x256 GEMMs per block via WMMA
  int wave = tid >> 5, lane = tid & 31, half = lane >> 4, l16 = lane & 15;
  for (int mt = wave; mt < 16; mt += 8) {
    const unsigned short* arow = mbuf + (mt * 16 + l16) * ESTR;
    bf16x16 afr[8];
#pragma unroll
    for (int kk = 0; kk < 8; ++kk) {
      bf16x8 lo = *(const bf16x8*)(arow + kk * 32 + half * 16);
      bf16x8 hi = *(const bf16x8*)(arow + kk * 32 + half * 16 + 8);
      bf16x16 a;
#pragma unroll
      for (int e = 0; e < 8; ++e) { a[e] = lo[e]; a[e + 8] = hi[e]; }
      afr[kk] = a;
    }
    // GEMM1: m = silu(m1 @ ew2 + eb2), in-place into mbuf (A frags are in regs)
    for (int nt = 0; nt < 16; ++nt) {
      int ncol = nt * 16 + l16;
      const unsigned short* wrow = ew2T + ncol * 256;
      if (nt < 15) __builtin_prefetch(wrow + 16 * 256, 0, 1);  // next N-tile row
      bf16x16 bfr[8];
#pragma unroll
      for (int kk = 0; kk < 8; ++kk)
        bfr[kk] = *(const bf16x16*)(wrow + kk * 32 + half * 16);
      f32x8 acc = {};
#pragma unroll
      for (int kk = 0; kk < 8; ++kk) acc = wmma_bf16(afr[kk], bfr[kk], acc);
      float bb = eb2[ncol];
#pragma unroll
      for (int r = 0; r < 8; ++r)
        mbuf[(mt * 16 + r + 8 * half) * ESTR + ncol] = f2bf(silu_f(acc[r] + bb));
    }
    // GEMM2: gate = silu(m @ cw1 + cb1) . cw2  (folded column reduction)
    bf16x16 afr2[8];
#pragma unroll
    for (int kk = 0; kk < 8; ++kk) {
      bf16x8 lo = *(const bf16x8*)(arow + kk * 32 + half * 16);
      bf16x8 hi = *(const bf16x8*)(arow + kk * 32 + half * 16 + 8);
      bf16x16 a;
#pragma unroll
      for (int e = 0; e < 8; ++e) { a[e] = lo[e]; a[e + 8] = hi[e]; }
      afr2[kk] = a;
    }
    float csum[8];
#pragma unroll
    for (int r = 0; r < 8; ++r) csum[r] = 0.f;
    for (int nt = 0; nt < 16; ++nt) {
      int ncol = nt * 16 + l16;
      const unsigned short* wrow = cw1T + ncol * 256;
      if (nt < 15) __builtin_prefetch(wrow + 16 * 256, 0, 1);
      bf16x16 bfr[8];
#pragma unroll
      for (int kk = 0; kk < 8; ++kk)
        bfr[kk] = *(const bf16x16*)(wrow + kk * 32 + half * 16);
      f32x8 acc = {};
#pragma unroll
      for (int kk = 0; kk < 8; ++kk) acc = wmma_bf16(afr2[kk], bfr[kk], acc);
      float bb = cb1[ncol], cw = cw2[ncol];
#pragma unroll
      for (int r = 0; r < 8; ++r) csum[r] += silu_f(acc[r] + bb) * cw;
    }
#pragma unroll
    for (int r = 0; r < 8; ++r) atomicAdd(&cacc[mt * 16 + r + 8 * half], csum[r]);
  }
  __syncthreads();

  // ---- magg_i += sum_j m   (col = tid)
  for (int i = 0; i < 16; ++i) {
    float s = 0.f;
    for (int j = 0; j < 16; ++j) s += bf2f(mbuf[(i * 16 + j) * ESTR + tid]);
    atomicAdd(&magg[(rowI + i) * 256 + tid], s);
  }
  // ---- coord contribution: sum_j (x_i - x_j) * c
  if (tid < 16) {
    float cb2v = cb2p[0];
    int i = tid;
    float sx = 0.f, sy = 0.f, sz = 0.f;
    for (int j = 0; j < 16; ++j) {
      float c = cacc[i * 16 + j] + cb2v;
      sx += (xi[i * 3] - xj[j * 3]) * c;
      sy += (xi[i * 3 + 1] - xj[j * 3 + 1]) * c;
      sz += (xi[i * 3 + 2] - xj[j * 3 + 2]) * c;
    }
    atomicAdd(&xacc[(rowI + i) * 3 + 0], sx);
    atomicAdd(&xacc[(rowI + i) * 3 + 1], sy);
    atomicAdd(&xacc[(rowI + i) * 3 + 2], sz);
  }
}

__global__ void zero_kernel(float* p, int n) {
  int i = blockIdx.x * blockDim.x + threadIdx.x;
  if (i < n) p[i] = 0.f;
}

__global__ void xupd_kernel(float* __restrict__ x, const float* __restrict__ xacc) {
  int i = blockIdx.x * 256 + threadIdx.x;
  if (i < ROWS * 3) x[i] += xacc[i] * (1.f / 256.f);   // mean over j
}

// Final H->3 projection
__global__ void proj_kernel(const float* __restrict__ o2, const float* __restrict__ w3,
                            const float* __restrict__ b3, float* __restrict__ out) {
  int idx = blockIdx.x * 256 + threadIdx.x;
  if (idx >= ROWS * 3) return;
  int row = idx / 3, c = idx % 3;
  float acc = b3[c];
  const float* orow = o2 + row * 256;
  for (int i = 0; i < 256; ++i) acc += orow[i] * w3[i * 3 + c];
  out[idx] = acc;
}

// ---------------------------------------------------------------------------
extern "C" void kernel_launch(void* const* d_in, const int* in_sizes, int n_in,
                              void* d_out, int out_size, void* d_ws, size_t ws_size,
                              hipStream_t stream) {
  (void)in_sizes; (void)n_in; (void)out_size; (void)ws_size;

  const float* coords = (const float*)d_in[0];
  const float* feats  = (const float*)d_in[1];
  const float* t_in   = (const float*)d_in[2];
  const float* pe_w1  = (const float*)d_in[3];
  const float* pe_b1  = (const float*)d_in[4];
  const float* pe_w2  = (const float*)d_in[5];
  const float* pe_b2  = (const float*)d_in[6];
  const float* pe_w3  = (const float*)d_in[7];
  const float* pe_b3  = (const float*)d_in[8];
  const float* te_w1  = (const float*)d_in[9];
  const float* te_b1  = (const float*)d_in[10];
  const float* te_w2  = (const float*)d_in[11];
  const float* te_b2  = (const float*)d_in[12];
  const float* ln_g   = (const float*)d_in[13];
  const float* ln_b   = (const float*)d_in[14];
  const float* ew1    = (const float*)d_in[15];
  const float* eb1    = (const float*)d_in[16];
  const float* ew2    = (const float*)d_in[17];
  const float* eb2    = (const float*)d_in[18];
  const float* cw1    = (const float*)d_in[19];
  const float* cb1    = (const float*)d_in[20];
  const float* cw2    = (const float*)d_in[21];
  const float* cb2    = (const float*)d_in[22];
  const float* nw1    = (const float*)d_in[23];
  const float* nb1    = (const float*)d_in[24];
  const float* nw2    = (const float*)d_in[25];
  const float* nb2    = (const float*)d_in[26];
  const float* dec_w1 = (const float*)d_in[27];
  const float* dec_b1 = (const float*)d_in[28];
  const float* dec_w2 = (const float*)d_in[29];
  const float* dec_b2 = (const float*)d_in[30];
  const float* dec_w3 = (const float*)d_in[31];
  const float* dec_b3 = (const float*)d_in[32];

  char* ws = (char*)d_ws;
  const size_t SZ = (size_t)ROWS * 256 * 4;   // 512 KB per activation buffer
  float* hbuf  = (float*)(ws + 0 * SZ);
  float* hn    = (float*)(ws + 1 * SZ);
  float* Abuf  = (float*)(ws + 2 * SZ);
  float* Bmbuf = (float*)(ws + 3 * SZ);
  float* hnN   = (float*)(ws + 4 * SZ);
  float* magg  = (float*)(ws + 5 * SZ);
  float* Ubuf  = (float*)(ws + 6 * SZ);
  float* o1    = (float*)(ws + 7 * SZ);
  float* o2    = (float*)(ws + 8 * SZ);
  float* xbuf  = (float*)(ws + 9 * SZ);                 // 6 KB
  float* xacc  = (float*)(ws + 9 * SZ + 8192);          // 6 KB
  unsigned short* wts = (unsigned short*)(ws + 9 * SZ + 16384);
  const size_t WMAT = 256 * 256;                        // bf16 elements per matrix
  auto WT = [&](int i) { return wts + (size_t)i * WMAT; };
  // matrix slots per layer l: 7l+{0:W1a,1:W1b,2:ew2,3:cw1,4:nw1a,5:nw1b,6:nw2};
  // 42:dec_w1, 43:dec_w2

  // ---- weight prep (bf16 transpose)
  for (int l = 0; l < LAYERS; ++l) {
    const float* ew1_l = ew1 + (size_t)l * 513 * 256;
    const float* nw1_l = nw1 + (size_t)l * 512 * 256;
    ct_kernel<<<256, 256, 0, stream>>>(ew1_l,               WT(7 * l + 0));
    ct_kernel<<<256, 256, 0, stream>>>(ew1_l + 256 * 256,   WT(7 * l + 1));
    ct_kernel<<<256, 256, 0, stream>>>(ew2 + (size_t)l * WMAT, WT(7 * l + 2));
    ct_kernel<<<256, 256, 0, stream>>>(cw1 + (size_t)l * WMAT, WT(7 * l + 3));
    ct_kernel<<<256, 256, 0, stream>>>(nw1_l,               WT(7 * l + 4));
    ct_kernel<<<256, 256, 0, stream>>>(nw1_l + 256 * 256,   WT(7 * l + 5));
    ct_kernel<<<256, 256, 0, stream>>>(nw2 + (size_t)l * WMAT, WT(7 * l + 6));
  }
  ct_kernel<<<256, 256, 0, stream>>>(dec_w1, WT(42));
  ct_kernel<<<256, 256, 0, stream>>>(dec_w2, WT(43));

  // ---- encoder + time embedding; x = coords
  enc_kernel<<<ROWS, 256, 0, stream>>>(coords, feats, t_in,
                                       pe_w1, pe_b1, pe_w2, pe_b2, pe_w3, pe_b3,
                                       te_w1, te_b1, te_w2, te_b2, hbuf, xbuf);

  const size_t ESMEM = 256 * ESTR * 2 + 256 * 4 + 256 * 4 + 48 * 4 + 48 * 4;

  for (int l = 0; l < LAYERS; ++l) {
    const float* ew1_l = ew1 + (size_t)l * 513 * 256;
    const float* w1c_l = ew1_l + 512 * 256;
    ln_kernel<<<ROWS, 256, 0, stream>>>(hbuf, ln_g + l * 256, ln_b + l * 256, hn);
    // per-node halves of the edge GEMM + node-update first half
    gemm_kernel<0, true, false, false><<<32, 256, 0, stream>>>(
        hn, WT(7 * l + 0), eb1 + l * 256, nullptr, nullptr, Abuf);
    gemm_kernel<0, false, false, false><<<32, 256, 0, stream>>>(
        hn, WT(7 * l + 1), nullptr, nullptr, nullptr, Bmbuf);
    gemm_kernel<0, true, false, false><<<32, 256, 0, stream>>>(
        hn, WT(7 * l + 4), nb1 + l * 256, nullptr, nullptr, hnN);
    zero_kernel<<<(ROWS * 256 + 255) / 256, 256, 0, stream>>>(magg, ROWS * 256);
    zero_kernel<<<6, 256, 0, stream>>>(xacc, ROWS * 3);
    // fused all-pairs edge MLP (dominant WMMA work)
    edge_kernel<<<dim3(16, 16, BB), 256, ESMEM, stream>>>(
        Abuf, Bmbuf, xbuf, w1c_l,
        WT(7 * l + 2), eb2 + l * 256,
        WT(7 * l + 3), cb1 + l * 256,
        cw2 + l * 256, cb2 + l,
        magg, xacc);
    xupd_kernel<<<6, 256, 0, stream>>>(xbuf, xacc);
    // node update: u = silu(hnN + magg@nw1b); h = h + u@nw2 + nb2
    gemm_kernel<2, false, true, false><<<32, 256, 0, stream>>>(
        magg, WT(7 * l + 5), nullptr, hnN, nullptr, Ubuf);
    gemm_kernel<0, true, true, false><<<32, 256, 0, stream>>>(
        Ubuf, WT(7 * l + 6), nb2 + l * 256, hbuf, nullptr, hbuf);
  }

  // ---- decoder
  gemm_kernel<1, true, false, false><<<32, 256, 0, stream>>>(
      hbuf, WT(42), dec_b1, nullptr, nullptr, o1);
  gemm_kernel<1, true, false, true><<<32, 256, 0, stream>>>(
      o1, WT(43), dec_b2, nullptr, hbuf, o2);
  proj_kernel<<<6, 256, 0, stream>>>(o2, dec_w3, dec_b3, (float*)d_out);
}
